// Attention_25563645346689
// MI455X (gfx1250) — matrix operand order repaired and checked by reading, not compiled, not run on hardware
//
#include <hip/hip_runtime.h>
#include <hip/hip_bf16.h>

// ---------------------------------------------------------------------------
// Relative-position attention (B=2,H=16,S=2048,D=64), fp32 in/out.
// d_out = [ output (BH*S*64) | p_attn (BH*S*S) ]  (reference returns a tuple)
//
// k0: build sin/cos table (fp32 + bf16 row-major + bf16 transposed) in ws
// k1: flash-style attention. Pass A: online softmax stats + O=P@V accum
//     (bf16 WMMA, fp32 acc) + far-band probability mass. Pass B: recompute
//     scores, write normalized p (non-temporal, 536MB stream).
//     Rel-K bias: per-q-row 257-entry band via WMMA (bias[q,t]=Q[q].table[t]).
// k2: rel-V term: out += Pd @ table_band + farL*table[0] + farR*table[256],
//     where Pd is the 257-wide diagonal band of p (gathered, WMMA'd).
// ---------------------------------------------------------------------------

typedef __attribute__((ext_vector_type(16))) __bf16          v16bf;
typedef __attribute__((ext_vector_type(8)))  float           v8f;
typedef __attribute__((ext_vector_type(16))) unsigned short  v16u;
typedef __attribute__((ext_vector_type(8)))  unsigned short  v8u;

#define S_LEN  2048
#define NBH    32
#define DHEAD  64
#define TROWS  288     // table rows padded (257 used, zeros beyond)
#define BIASW  272     // bias band width padded to 17*16
#define SCALE  0.125f  // 1/sqrt(64)

// native bf16 convert (v_cvt class ops instead of manual bit twiddling)
__device__ __forceinline__ unsigned short f2bf(float f) {
  __bf16 h = (__bf16)f;
  return __builtin_bit_cast(unsigned short, h);
}
__device__ __forceinline__ float bf2f(unsigned short h) {
  unsigned u = ((unsigned)h) << 16;
  return __builtin_bit_cast(float, u);
}
__device__ __forceinline__ int iclamp(int v, int lo, int hi) {
  return v < lo ? lo : (v > hi ? hi : v);
}

__device__ __forceinline__ v8f wmma_bf16(v16bf a, v16bf b, v8f c) {
  return __builtin_amdgcn_wmma_f32_16x16x32_bf16(false, a, false, b, (short)0, c,
                                                 false, false);
}

// A operand (16x32 bf16): lane&15 = row M; kb = lane<16 ? 0 : 8;
// lane holds K = {kb..kb+7} U {kb+16..kb+23}.
__device__ __forceinline__ v16bf loadA_f32(const float* rowBase, int rowStride, int lane) {
  int r  = lane & 15;
  int kb = (lane & 16) ? 8 : 0;
  const float* p = rowBase + r * rowStride + kb;
  v16u u;
#pragma unroll
  for (int j = 0; j < 8; ++j) { u[j] = f2bf(p[j]); u[8 + j] = f2bf(p[16 + j]); }
  return __builtin_bit_cast(v16bf, u);
}
__device__ __forceinline__ v16bf loadA_lds(const unsigned short* base, int rowStride, int lane) {
  int r  = lane & 15;
  int kb = (lane & 16) ? 8 : 0;
  const unsigned short* p = base + r * rowStride + kb;
  v8u lo = *(const v8u*)(p);
  v8u hi = *(const v8u*)(p + 16);
  v16u u;
#pragma unroll
  for (int j = 0; j < 8; ++j) { u[j] = lo[j]; u[8 + j] = hi[j]; }
  return __builtin_bit_cast(v16bf, u);
}
// B operand (32x16 bf16): column N = lane&15; kb = lane<16 ? 0 : 8;
// lane holds K = {kb..kb+7} U {kb+16..kb+23}.  colBase points at (K=0) of this lane's column.
__device__ __forceinline__ v16bf loadB_u16(const unsigned short* colBase, int lane) {
  int kb = (lane & 16) ? 8 : 0;
  v8u lo = *(const v8u*)(colBase + kb);
  v8u hi = *(const v8u*)(colBase + kb + 16);
  v16u u;
#pragma unroll
  for (int j = 0; j < 8; ++j) { u[j] = lo[j]; u[8 + j] = hi[j]; }
  return __builtin_bit_cast(v16bf, u);
}

// per-row reductions inside a 16-lane group (rows split lanes 0-15 / 16-31)
__device__ __forceinline__ float redmax16(float v) {
#pragma unroll
  for (int m = 1; m <= 8; m <<= 1) v = fmaxf(v, __shfl_xor(v, m, 32));
  return v;
}
__device__ __forceinline__ float redsum16(float v) {
#pragma unroll
  for (int m = 1; m <= 8; m <<= 1) v += __shfl_xor(v, m, 32);
  return v;
}

// ---------------------------------------------------------------------------
__global__ void k0_table(float* tabf, unsigned short* tabb, unsigned short* tabT) {
  int idx = blockIdx.x * blockDim.x + threadIdx.x;
  if (idx >= TROWS * DHEAD) return;
  int t = idx >> 6, d = idx & 63;
  float v = 0.f;
  if (t <= 256) {
    float e   = (float)(d & ~1);
    float div = __expf(e * (-9.210340371976184f / 64.f));
    float ang = (float)t * div;
    v = (d & 1) ? __cosf(ang) : __sinf(ang);
  }
  tabf[idx] = v;
  unsigned short b = f2bf(v);
  tabb[t * DHEAD + d] = b;
  tabT[d * TROWS + t] = b;
}

// ---------------------------------------------------------------------------
__global__ __launch_bounds__(128) void k1_attn(
    const float* __restrict__ Q, const float* __restrict__ K,
    const float* __restrict__ V, const int* __restrict__ oneDirPtr,
    const unsigned short* __restrict__ tabb,
    float* __restrict__ outO, float* __restrict__ outP,
    float* __restrict__ farLg, float* __restrict__ farRg) {
  __shared__ unsigned short s_kt[32 * 64];       // K tile, [kpos][d]
  __shared__ unsigned short s_vt[64 * 48];       // V tile transposed [d][kpos], padded
  __shared__ unsigned short s_bias[64 * BIASW];  // bias band per q row (bf16)
  __shared__ unsigned short s_p[4][16 * 32];     // per-wave P staging

  const int tid  = threadIdx.x;
  const int lane = tid & 31;
  const int w    = tid >> 5;
  const int bh   = blockIdx.x >> 5;
  const int q0   = (blockIdx.x & 31) * 64;
  const int causal = (*oneDirPtr != 0);

  const int hi8   = (lane & 16) ? 8 : 0;
  const int colb  = lane & 15;
  const int qrowW = q0 + w * 16;

  const float* Qb = Q + (size_t)bh * S_LEN * DHEAD;
  const float* Kb = K + (size_t)bh * S_LEN * DHEAD;
  const float* Vb = V + (size_t)bh * S_LEN * DHEAD;

  // Q A-operands (two d-chunks of 32)
  v16bf Aq0 = loadA_f32(Qb + (size_t)qrowW * DHEAD + 0, DHEAD, lane);
  v16bf Aq1 = loadA_f32(Qb + (size_t)qrowW * DHEAD + 32, DHEAD, lane);

  // ---- bias band via WMMA: bias[r][t] = Q[r] . table[t] ----
#pragma unroll 1
  for (int nt = 0; nt < 17; ++nt) {
    v8f acc = {};
    const unsigned short* cb = tabb + (size_t)(nt * 16 + colb) * DHEAD;
    acc = wmma_bf16(Aq0, loadB_u16(cb + 0, lane), acc);
    acc = wmma_bf16(Aq1, loadB_u16(cb + 32, lane), acc);
#pragma unroll
    for (int i = 0; i < 8; ++i)
      s_bias[(w * 16 + i + hi8) * BIASW + nt * 16 + colb] = f2bf(acc[i]);
  }
  asm volatile("s_wait_dscnt 0" ::: "memory");

  float mrun[8], lrun[8], fL[8], fR[8];
#pragma unroll
  for (int i = 0; i < 8; ++i) { mrun[i] = -1e30f; lrun[i] = 0.f; fL[i] = 0.f; fR[i] = 0.f; }
  v8f O0 = {}, O1 = {}, O2 = {}, O3 = {};

  // =================== Pass A: stats + O accumulation ===================
  const int ktA = causal ? ((q0 + 63) >> 5) : (S_LEN / 32 - 1);
#pragma unroll 1
  for (int kt = 0; kt <= ktA; ++kt) {
    __syncthreads();
#pragma unroll
    for (int e = 0; e < 16; ++e) {
      int lin = tid + e * 128;
      int d = lin & 63, kp = lin >> 6;
      size_t g = (size_t)(kt * 32 + kp) * DHEAD + d;
      s_kt[kp * 64 + d] = f2bf(Kb[g]);
      s_vt[d * 48 + kp] = f2bf(Vb[g]);
    }
    if (kt < ktA) {  // prefetch next tile (global_prefetch_b8)
      __builtin_prefetch(Kb + (size_t)((kt + 1) * 32 + (tid >> 2)) * DHEAD, 0, 1);
      __builtin_prefetch(Vb + (size_t)((kt + 1) * 32 + (tid >> 2)) * DHEAD, 0, 1);
    }
    __syncthreads();

    // S = Q K^T  (two 16-col halves, two 32-d chunks)
    v8f s0 = {}, s1 = {};
    {
      const unsigned short* cb0 = s_kt + (0 * 16 + colb) * 64;
      const unsigned short* cb1 = s_kt + (1 * 16 + colb) * 64;
      s0 = wmma_bf16(Aq0, loadB_u16(cb0 + 0, lane), s0);
      s0 = wmma_bf16(Aq1, loadB_u16(cb0 + 32, lane), s0);
      s1 = wmma_bf16(Aq0, loadB_u16(cb1 + 0, lane), s1);
      s1 = wmma_bf16(Aq1, loadB_u16(cb1 + 32, lane), s1);
    }

    float p0[8], p1[8];
#pragma unroll
    for (int i = 0; i < 8; ++i) {
      int qg = qrowW + i + hi8;
      int k0g = kt * 32 + colb, k1g = k0g + 16;
      int t0 = iclamp(k0g - qg, -128, 128) + 128;
      int t1 = iclamp(k1g - qg, -128, 128) + 128;
      int br = (w * 16 + i + hi8) * BIASW;
      float v0 = (s0[i] + bf2f(s_bias[br + t0])) * SCALE;
      float v1 = (s1[i] + bf2f(s_bias[br + t1])) * SCALE;
      if (causal && k0g > qg) v0 = -3.0e38f;
      if (causal && k1g > qg) v1 = -3.0e38f;
      p0[i] = v0; p1[i] = v1;
    }

#pragma unroll
    for (int i = 0; i < 8; ++i) {
      float tmax = redmax16(fmaxf(p0[i], p1[i]));
      float mnew = fmaxf(mrun[i], tmax);
      float corr = __expf(mrun[i] - mnew);
      float e0 = __expf(p0[i] - mnew);
      float e1 = __expf(p1[i] - mnew);
      p0[i] = e0; p1[i] = e1;
      lrun[i] = lrun[i] * corr + redsum16(e0 + e1);
      int qg = qrowW + i + hi8;
      int k0g = kt * 32 + colb, k1g = k0g + 16;
      float fl = ((k0g < qg - 128) ? e0 : 0.f) + ((k1g < qg - 128) ? e1 : 0.f);
      float fr = ((k0g > qg + 128) ? e0 : 0.f) + ((k1g > qg + 128) ? e1 : 0.f);
      fL[i] = fL[i] * corr + redsum16(fl);
      fR[i] = fR[i] * corr + redsum16(fr);
      mrun[i] = mnew;
      O0[i] *= corr; O1[i] *= corr; O2[i] *= corr; O3[i] *= corr;
    }

    // P -> LDS (C-layout scatter), then O += P @ V
#pragma unroll
    for (int i = 0; i < 8; ++i) {
      s_p[w][(i + hi8) * 32 + colb]      = f2bf(p0[i]);
      s_p[w][(i + hi8) * 32 + colb + 16] = f2bf(p1[i]);
    }
    asm volatile("s_wait_dscnt 0" ::: "memory");
    v16bf Pa = loadA_lds(&s_p[w][0], 32, lane);
    O0 = wmma_bf16(Pa, loadB_u16(s_vt + (0 * 16 + colb) * 48, lane), O0);
    O1 = wmma_bf16(Pa, loadB_u16(s_vt + (1 * 16 + colb) * 48, lane), O1);
    O2 = wmma_bf16(Pa, loadB_u16(s_vt + (2 * 16 + colb) * 48, lane), O2);
    O3 = wmma_bf16(Pa, loadB_u16(s_vt + (3 * 16 + colb) * 48, lane), O3);
  }

  float invl[8];
#pragma unroll
  for (int i = 0; i < 8; ++i) invl[i] = (lrun[i] > 0.f) ? (1.f / lrun[i]) : 0.f;

  // =================== Pass B: write normalized p (NT stream) ===========
#pragma unroll 1
  for (int kt = 0; kt < S_LEN / 32; ++kt) {
    bool needWG = (!causal) || (kt * 32 <= q0 + 63);
    __syncthreads();
    if (needWG) {
#pragma unroll
      for (int e = 0; e < 16; ++e) {
        int lin = tid + e * 128;
        int d = lin & 63, kp = lin >> 6;
        s_kt[kp * 64 + d] = f2bf(Kb[(size_t)(kt * 32 + kp) * DHEAD + d]);
      }
    }
    __syncthreads();

    if (causal && (kt * 32 > qrowW + 15)) {  // fully masked for this wave
#pragma unroll
      for (int i = 0; i < 8; ++i) {
        size_t r = ((size_t)bh * S_LEN + qrowW + i + hi8) * S_LEN + kt * 32;
        __builtin_nontemporal_store(0.f, &outP[r + colb]);
        __builtin_nontemporal_store(0.f, &outP[r + colb + 16]);
      }
      continue;
    }

    v8f s0 = {}, s1 = {};
    const unsigned short* cb0 = s_kt + (colb)*64;
    const unsigned short* cb1 = s_kt + (16 + colb) * 64;
    s0 = wmma_bf16(Aq0, loadB_u16(cb0 + 0, lane), s0);
    s0 = wmma_bf16(Aq1, loadB_u16(cb0 + 32, lane), s0);
    s1 = wmma_bf16(Aq0, loadB_u16(cb1 + 0, lane), s1);
    s1 = wmma_bf16(Aq1, loadB_u16(cb1 + 32, lane), s1);

#pragma unroll
    for (int i = 0; i < 8; ++i) {
      int qg = qrowW + i + hi8;
      int k0g = kt * 32 + colb, k1g = k0g + 16;
      int t0 = iclamp(k0g - qg, -128, 128) + 128;
      int t1 = iclamp(k1g - qg, -128, 128) + 128;
      int br = (w * 16 + i + hi8) * BIASW;
      float v0 = (s0[i] + bf2f(s_bias[br + t0])) * SCALE;
      float v1 = (s1[i] + bf2f(s_bias[br + t1])) * SCALE;
      if (causal && k0g > qg) v0 = -3.0e38f;
      if (causal && k1g > qg) v1 = -3.0e38f;
      size_t r = ((size_t)bh * S_LEN + qg) * S_LEN + kt * 32;
      __builtin_nontemporal_store(__expf(v0 - mrun[i]) * invl[i], &outP[r + colb]);
      __builtin_nontemporal_store(__expf(v1 - mrun[i]) * invl[i], &outP[r + colb + 16]);
    }
  }

  // ---- O/l, far mass ----
#pragma unroll
  for (int i = 0; i < 8; ++i) {
    int qg = qrowW + i + hi8;
    size_t ro = ((size_t)bh * S_LEN + qg) * DHEAD;
    outO[ro + 0 * 16 + colb] = O0[i] * invl[i];
    outO[ro + 1 * 16 + colb] = O1[i] * invl[i];
    outO[ro + 2 * 16 + colb] = O2[i] * invl[i];
    outO[ro + 3 * 16 + colb] = O3[i] * invl[i];
  }
  if (colb == 0) {
#pragma unroll
    for (int i = 0; i < 8; ++i) {
      int qg = qrowW + i + hi8;
      farLg[bh * S_LEN + qg] = fL[i] * invl[i];
      farRg[bh * S_LEN + qg] = fR[i] * invl[i];
    }
  }
}

// ---------------------------------------------------------------------------
__global__ __launch_bounds__(128) void k2_relv(
    const float* __restrict__ outPc, float* __restrict__ outO,
    const unsigned short* __restrict__ tabT, const float* __restrict__ tabf,
    const float* __restrict__ farLg, const float* __restrict__ farRg) {
  __shared__ unsigned short s_pd[4][16 * 32];
  const int tid = threadIdx.x, lane = tid & 31, w = tid >> 5;
  const int hi8 = (lane & 16) ? 8 : 0, colb = lane & 15;
  const int row0 = blockIdx.x * 64 + w * 16;  // global (bh,q) row
  const int bh = row0 / S_LEN;
  const int q0 = row0 % S_LEN;

  v8f acc[4] = {};
#pragma unroll 1
  for (int c = 0; c < 9; ++c) {
    // gather diagonal band chunk: Pd[r][j] = p[q0+r][q0+r + c*32+j - 128]
    // iteration e covers row r=e across all 32 lanes (coalesced 128B)
#pragma unroll
    for (int e = 0; e < 16; ++e) {
      int r = e, j = lane;
      int t = c * 32 + j;
      int q = q0 + r;
      int k = q + t - 128;
      float v = 0.f;
      if (t <= 256 && k >= 0 && k < S_LEN)
        v = outPc[((size_t)bh * S_LEN + q) * S_LEN + k];
      s_pd[w][r * 32 + j] = f2bf(v);
    }
    asm volatile("s_wait_dscnt 0" ::: "memory");
    v16bf Pa = loadA_lds(&s_pd[w][0], 32, lane);
#pragma unroll
    for (int nt = 0; nt < 4; ++nt)
      acc[nt] = wmma_bf16(
          Pa, loadB_u16(tabT + (size_t)(nt * 16 + colb) * TROWS + c * 32, lane),
          acc[nt]);
  }

  float fl[8], fr[8];
#pragma unroll
  for (int i = 0; i < 8; ++i) {
    int gq = row0 + i + hi8;
    fl[i] = farLg[gq];
    fr[i] = farRg[gq];
  }
#pragma unroll
  for (int nt = 0; nt < 4; ++nt) {
    int d = nt * 16 + colb;
    float t0 = tabf[d];
    float t256 = tabf[256 * DHEAD + d];
#pragma unroll
    for (int i = 0; i < 8; ++i) {
      int gq = row0 + i + hi8;
      size_t o = (size_t)gq * DHEAD + d;
      outO[o] = outO[o] + acc[nt][i] + fl[i] * t0 + fr[i] * t256;
    }
  }
}

// ---------------------------------------------------------------------------
extern "C" void kernel_launch(void* const* d_in, const int* in_sizes, int n_in,
                              void* d_out, int out_size, void* d_ws, size_t ws_size,
                              hipStream_t stream) {
  (void)in_sizes; (void)n_in; (void)out_size; (void)ws_size;
  const float* Q = (const float*)d_in[0];
  const float* K = (const float*)d_in[1];
  const float* V = (const float*)d_in[2];
  const int* od  = (const int*)d_in[3];
  float* out  = (float*)d_out;
  float* pOut = out + (size_t)NBH * S_LEN * DHEAD;

  char* ws = (char*)d_ws;
  float*          tabf = (float*)ws;                          // 73728 B
  unsigned short* tabb = (unsigned short*)(ws + 73728);       // 36864 B
  unsigned short* tabT = (unsigned short*)(ws + 110592);      // 36864 B
  float*          farL = (float*)(ws + 147456);               // 262144 B
  float*          farR = (float*)(ws + 147456 + 262144);      // 262144 B

  k0_table<<<(TROWS * DHEAD + 255) / 256, 256, 0, stream>>>(tabf, tabb, tabT);
  k1_attn<<<NBH * (S_LEN / 64), 128, 0, stream>>>(Q, K, V, od, tabb, out, pOut,
                                                  farL, farR);
  k2_relv<<<(NBH * S_LEN) / 64, 128, 0, stream>>>(pOut, out, tabT, tabf, farL,
                                                  farR);
}
